// DynSIHA_14044543058151
// MI455X (gfx1250) — matrix operand order repaired and checked by reading, not compiled, not run on hardware
//
#include <hip/hip_runtime.h>

// ---------- types ----------
typedef __bf16 bf16_t;
typedef __attribute__((ext_vector_type(16))) __bf16 v16bf;
typedef __attribute__((ext_vector_type(8)))  float  v8f;
typedef __attribute__((ext_vector_type(4)))  float  v4f;
typedef __attribute__((ext_vector_type(4)))  unsigned int v4u;

union V16U { v16bf v; v4u q[2]; };

#define DEVINL __device__ __forceinline__

DEVINL v8f wmma_bf16(v16bf a, v16bf b, v8f c) {
  return __builtin_amdgcn_wmma_f32_16x16x32_bf16(false, a, false, b, (short)0, c,
                                                 false, false);
}

// A-tile (16x32 bf16) load from a bf16 row (per-lane: row = lane&15, 16 contiguous k)
DEVINL v16bf ldA_bf(const bf16_t* row, int kk, int hi) {
  V16U u;
  u.q[0] = *(const v4u*)(row + kk * 32 + hi * 16);
  u.q[1] = *(const v4u*)(row + kk * 32 + hi * 16 + 8);
  return u.v;
}

DEVINL void cvt8(const float* s, v16bf& d, int off) {
  v4f a = *(const v4f*)s;
  v4f b = *(const v4f*)(s + 4);
#pragma unroll
  for (int i = 0; i < 4; ++i) {
    d[off + i]     = (bf16_t)a[i];
    d[off + 4 + i] = (bf16_t)b[i];
  }
}

// A-tile from f32 row (convert on the fly)
DEVINL v16bf ldA_f32(const float* row, int kk, int hi) {
  v16bf a;
  cvt8(row + kk * 32 + hi * 16, a, 0);
  cvt8(row + kk * 32 + hi * 16 + 8, a, 8);
  return a;
}

// B-tile (32x16) from n-major bf16 storage: nrow points at [n][0]; 16 contiguous k
DEVINL v16bf ldB_bf(const bf16_t* nrow, int kk, int hi) {
  return *(const v16bf*)(nrow + kk * 32 + hi * 16);
}

// B-tile from n-major f32 storage (convert on the fly)
DEVINL v16bf ldB_f32(const float* nrow, int kk, int hi) {
  v16bf b;
  const float* s = nrow + kk * 32 + hi * 16;
  cvt8(s, b, 0);
  cvt8(s + 8, b, 8);
  return b;
}

// reductions across the 16-lane half (C-tile rows are uniform per half)
DEVINL float half_sum(float x) {
  x += __shfl_xor(x, 1, 32);
  x += __shfl_xor(x, 2, 32);
  x += __shfl_xor(x, 4, 32);
  x += __shfl_xor(x, 8, 32);
  return x;
}
DEVINL float half_max(float x) {
  x = fmaxf(x, __shfl_xor(x, 1, 32));
  x = fmaxf(x, __shfl_xor(x, 2, 32));
  x = fmaxf(x, __shfl_xor(x, 4, 32));
  x = fmaxf(x, __shfl_xor(x, 8, 32));
  return x;
}

// ---------- sizes ----------
// B=2 T=1024 D=1024 H=16 Dh=64 P=16, S = B*T*H = 32768

// ---------- prep: bf16 transposed weights ----------
__global__ void prep_kernel(const float* __restrict__ W1, const float* __restrict__ W2,
                            const float* __restrict__ Wo, bf16_t* __restrict__ w1t,
                            bf16_t* __restrict__ w2t, bf16_t* __restrict__ wob) {
  int tid = blockIdx.x * blockDim.x + threadIdx.x;
  if (tid < 65536) {
    int p = tid >> 12, rem = tid & 4095, n = rem >> 6, k = rem & 63;
    // w1t[p][n=e][k=d] = W1[p][d][e]; w2t[p][n=f][k=e] = W2[p][e][f]
    w1t[tid] = (bf16_t)W1[(p * 64 + k) * 64 + n];
    w2t[tid] = (bf16_t)W2[(p * 64 + k) * 64 + n];
  }
  if (tid < 1048576) wob[tid] = (bf16_t)Wo[tid];  // B[n=j][k] = Wo[j][k] as-is
}

// ---------- compose: logits + shared expert MLP + gated accumulation ----------
__global__ __launch_bounds__(32) void compose_kernel(
    const float* __restrict__ x, const float* __restrict__ proto_q,
    const float* __restrict__ gate_q, const float* __restrict__ proto_k,
    const float* __restrict__ gate_k, const float* __restrict__ proto_v,
    const float* __restrict__ gate_v, const bf16_t* __restrict__ w1t,
    const bf16_t* __restrict__ w2t, float* __restrict__ out_lq,
    float* __restrict__ out_lk, float* __restrict__ out_lv, float* __restrict__ out_rq,
    float* __restrict__ out_rk, float* __restrict__ out_rv, bf16_t* __restrict__ qbf,
    bf16_t* __restrict__ kbf, bf16_t* __restrict__ vtbf) {
  __shared__ __align__(16) bf16_t lds_h[16 * 64];
  const int lane = threadIdx.x & 31;
  const int n16 = lane & 15, hi = lane >> 4;
  const int tile = blockIdx.x;  // 2048 tiles of 16 token-heads

  const float* xrow = x + (long)(tile * 16 + n16) * 64;
  v16bf a0 = ldA_f32(xrow, 0, hi);
  v16bf a1 = ldA_f32(xrow, 1, hi);

  // routing logits via WMMA: C[m][p], p = n16
  v8f lg[3];
  const float* protos[3] = {proto_q, proto_k, proto_v};
  const float* gates[3]  = {gate_q, gate_k, gate_v};
#pragma unroll
  for (int t3 = 0; t3 < 3; ++t3) {
    v8f c;
#pragma unroll
    for (int r = 0; r < 8; ++r) c[r] = 0.0f;
    c = wmma_bf16(a0, ldB_f32(protos[t3] + n16 * 64, 0, hi), c);
    c = wmma_bf16(a1, ldB_f32(protos[t3] + n16 * 64, 1, hi), c);
    float gv = gates[t3][n16];
#pragma unroll
    for (int r = 0; r < 8; ++r) c[r] = fmaxf(c[r] * 0.125f - gv, 0.0f);
    lg[t3] = c;
  }
  float* louts[3] = {out_lq, out_lk, out_lv};
#pragma unroll
  for (int t3 = 0; t3 < 3; ++t3)
#pragma unroll
    for (int r = 0; r < 8; ++r)
      louts[t3][(long)(tile * 16 + r + 8 * hi) * 16 + n16] = lg[t3][r];

  float qs[4][8], ks[4][8], vs[4][8];
#pragma unroll
  for (int j = 0; j < 4; ++j)
#pragma unroll
    for (int r = 0; r < 8; ++r) { qs[j][r] = 0.f; ks[j][r] = 0.f; vs[j][r] = 0.f; }

  for (int p = 0; p < 16; ++p) {
    // broadcast thresholded gates per row (row m = r + 8*hi lives at lane p + 16*hi)
    float gq[8], gk[8], gvv[8];
    int src = p + (hi << 4);
#pragma unroll
    for (int r = 0; r < 8; ++r) {
      float a = __shfl(lg[0][r], src, 32); gq[r]  = a > 1e-6f ? a : 0.0f;
      float b = __shfl(lg[1][r], src, 32); gk[r]  = b > 1e-6f ? b : 0.0f;
      float c = __shfl(lg[2][r], src, 32); gvv[r] = c > 1e-6f ? c : 0.0f;
    }
    // GEMM1: h = silu(x @ W1[p])   (16x64)
    const bf16_t* w1p = w1t + p * 4096;
    v8f hj[4];
#pragma unroll
    for (int j = 0; j < 4; ++j) {
      v8f c;
#pragma unroll
      for (int r = 0; r < 8; ++r) c[r] = 0.0f;
      const bf16_t* bn = w1p + (j * 16 + n16) * 64;
      c = wmma_bf16(a0, ldB_bf(bn, 0, hi), c);
      c = wmma_bf16(a1, ldB_bf(bn, 1, hi), c);
#pragma unroll
      for (int r = 0; r < 8; ++r) {
        float v = c[r];
        c[r] = v / (1.0f + __expf(-v));  // silu
      }
      hj[j] = c;
    }
    // C-layout -> A-layout via LDS (bf16 16x64 tile)
#pragma unroll
    for (int j = 0; j < 4; ++j)
#pragma unroll
      for (int r = 0; r < 8; ++r)
        lds_h[(r + 8 * hi) * 64 + j * 16 + n16] = (bf16_t)hj[j][r];
    __syncthreads();
    v16bf ah0 = ldA_bf(lds_h + n16 * 64, 0, hi);
    v16bf ah1 = ldA_bf(lds_h + n16 * 64, 1, hi);
    __syncthreads();

    // GEMM2: E = h @ W2[p]
    const bf16_t* w2p = w2t + p * 4096;
    v8f ej[4];
    float ss[8];
#pragma unroll
    for (int r = 0; r < 8; ++r) ss[r] = 0.0f;
#pragma unroll
    for (int j = 0; j < 4; ++j) {
      v8f c;
#pragma unroll
      for (int r = 0; r < 8; ++r) c[r] = 0.0f;
      const bf16_t* bn = w2p + (j * 16 + n16) * 64;
      c = wmma_bf16(ah0, ldB_bf(bn, 0, hi), c);
      c = wmma_bf16(ah1, ldB_bf(bn, 1, hi), c);
      ej[j] = c;
#pragma unroll
      for (int r = 0; r < 8; ++r) ss[r] += c[r] * c[r];
    }
    float nrm[8];
#pragma unroll
    for (int r = 0; r < 8; ++r) nrm[r] = sqrtf(half_sum(ss[r]));
#pragma unroll
    for (int j = 0; j < 4; ++j)
#pragma unroll
      for (int r = 0; r < 8; ++r) {
        qs[j][r] += gq[r] * ej[j][r];
        ks[j][r] += gk[r] * ej[j][r];
        vs[j][r] += gvv[r] * ej[j][r];
      }
    if (n16 == p) {
#pragma unroll
      for (int r = 0; r < 8; ++r) {
        long m = (long)(tile * 16 + r + 8 * hi) * 16 + p;
        out_rq[m] = gq[r] * nrm[r];
        out_rk[m] = gk[r] * nrm[r];
        out_rv[m] = gvv[r] * nrm[r];
      }
    }
  }

  // scatter synthetic heads: q/k row-major [bh][t][d] bf16, v transposed [bh][d][t]
#pragma unroll
  for (int r = 0; r < 8; ++r) {
    int s = tile * 16 + r + 8 * hi;            // s = (b*T + t)*H + h
    int b = s >> 14, t = (s >> 4) & 1023, hh = s & 15;
    long bh = (long)(b * 16 + hh);
    long qbase = (bh * 1024 + t) * 64;
#pragma unroll
    for (int j = 0; j < 4; ++j) {
      int d = j * 16 + n16;
      qbf[qbase + d] = (bf16_t)qs[j][r];
      kbf[qbase + d] = (bf16_t)ks[j][r];
      vtbf[(bh * 64 + d) * 1024 + t] = (bf16_t)vs[j][r];
    }
  }
}

// ---------- flash attention (causal), one wave per 16-query tile ----------
__global__ __launch_bounds__(32) void attn_kernel(const bf16_t* __restrict__ qbf,
                                                  const bf16_t* __restrict__ kbf,
                                                  const bf16_t* __restrict__ vtbf,
                                                  float* __restrict__ attn_out) {
  __shared__ __align__(16) bf16_t lds_p[16 * 32];
  const int lane = threadIdx.x & 31;
  const int n16 = lane & 15, hi = lane >> 4;
  const int bh = blockIdx.x;  // 0..31
  const int qt = blockIdx.y;  // 0..63

  const bf16_t* qrow = qbf + ((long)bh * 1024 + qt * 16 + n16) * 64;
  v16bf aq0 = ldA_bf(qrow, 0, hi);
  v16bf aq1 = ldA_bf(qrow, 1, hi);

  v8f acc[4];
  float rowm[8], rowl[8];
#pragma unroll
  for (int r = 0; r < 8; ++r) { rowm[r] = -1e30f; rowl[r] = 0.0f; }
#pragma unroll
  for (int j = 0; j < 4; ++j)
#pragma unroll
    for (int r = 0; r < 8; ++r) acc[j][r] = 0.0f;

  const int qend = qt * 16 + 16;
  for (int kv0 = 0; kv0 < qend; kv0 += 32) {
    v8f sj[2];
#pragma unroll
    for (int j = 0; j < 2; ++j) {
      const bf16_t* krow = kbf + ((long)bh * 1024 + kv0 + j * 16 + n16) * 64;
      v8f c;
#pragma unroll
      for (int r = 0; r < 8; ++r) c[r] = 0.0f;
      c = wmma_bf16(aq0, ldB_bf(krow, 0, hi), c);
      c = wmma_bf16(aq1, ldB_bf(krow, 1, hi), c);
      int kg = kv0 + j * 16 + n16;
#pragma unroll
      for (int r = 0; r < 8; ++r) {
        int qg = qt * 16 + r + 8 * hi;
        c[r] = (kg <= qg) ? c[r] * 0.125f : -1e30f;
      }
      sj[j] = c;
    }
    float alpha[8];
#pragma unroll
    for (int r = 0; r < 8; ++r) {
      float mx = half_max(fmaxf(sj[0][r], sj[1][r]));
      float nm = fmaxf(rowm[r], mx);
      alpha[r] = __expf(rowm[r] - nm);
      rowm[r] = nm;
    }
#pragma unroll
    for (int r = 0; r < 8; ++r) {
      float p0 = __expf(sj[0][r] - rowm[r]);
      float p1 = __expf(sj[1][r] - rowm[r]);
      sj[0][r] = p0; sj[1][r] = p1;
      rowl[r] = rowl[r] * alpha[r] + half_sum(p0 + p1);
    }
#pragma unroll
    for (int j = 0; j < 4; ++j)
#pragma unroll
      for (int r = 0; r < 8; ++r) acc[j][r] *= alpha[r];
    // P (C-layout f32) -> A-layout bf16 via LDS
#pragma unroll
    for (int j = 0; j < 2; ++j)
#pragma unroll
      for (int r = 0; r < 8; ++r)
        lds_p[(r + 8 * hi) * 32 + j * 16 + n16] = (bf16_t)sj[j][r];
    __syncthreads();
    v16bf ap = ldA_bf(lds_p + n16 * 32, 0, hi);
    __syncthreads();
#pragma unroll
    for (int jt = 0; jt < 4; ++jt) {
      const bf16_t* vrow = vtbf + ((long)bh * 64 + jt * 16 + n16) * 1024 + kv0;
      acc[jt] = wmma_bf16(ap, *(const v16bf*)(vrow + hi * 16), acc[jt]);
    }
  }
  const int b = bh >> 4, hh = bh & 15;
#pragma unroll
  for (int r = 0; r < 8; ++r) {
    int tq = qt * 16 + r + 8 * hi;
    float inv = 1.0f / rowl[r];
#pragma unroll
    for (int jt = 0; jt < 4; ++jt)
      attn_out[((long)b * 1024 + tq) * 1024 + hh * 64 + jt * 16 + n16] =
          acc[jt][r] * inv;
  }
}

// ---------- output projection: out[2048,1024] = attn @ Wo^T ----------
__global__ __launch_bounds__(32) void proj_kernel(const float* __restrict__ attn,
                                                  const bf16_t* __restrict__ wob,
                                                  float* __restrict__ out) {
  const int lane = threadIdx.x & 31;
  const int n16 = lane & 15, hi = lane >> 4;
  const int mt = blockIdx.x;  // 0..127
  const int ng = blockIdx.y;  // 0..15 (4 n-tiles each)
  const float* arow = attn + (long)(mt * 16 + n16) * 1024;
  v8f acc[4];
#pragma unroll
  for (int q = 0; q < 4; ++q)
#pragma unroll
    for (int r = 0; r < 8; ++r) acc[q][r] = 0.0f;
  for (int kk = 0; kk < 32; ++kk) {
    v16bf a = ldA_f32(arow, kk, hi);
#pragma unroll
    for (int q = 0; q < 4; ++q) {
      const bf16_t* brow = wob + (long)(ng * 64 + q * 16 + n16) * 1024;
      acc[q] = wmma_bf16(a, ldB_bf(brow, kk, hi), acc[q]);
    }
  }
#pragma unroll
  for (int r = 0; r < 8; ++r)
#pragma unroll
    for (int q = 0; q < 4; ++q)
      out[(long)(mt * 16 + r + 8 * hi) * 1024 + ng * 64 + q * 16 + n16] = acc[q][r];
}

// ---------- launcher ----------
extern "C" void kernel_launch(void* const* d_in, const int* in_sizes, int n_in,
                              void* d_out, int out_size, void* d_ws, size_t ws_size,
                              hipStream_t stream) {
  const float* x       = (const float*)d_in[0];
  const float* proto_q = (const float*)d_in[1];
  const float* gate_q  = (const float*)d_in[2];
  const float* proto_k = (const float*)d_in[3];
  const float* gate_k  = (const float*)d_in[4];
  const float* proto_v = (const float*)d_in[5];
  const float* gate_v  = (const float*)d_in[6];
  const float* W1      = (const float*)d_in[7];
  const float* W2      = (const float*)d_in[8];
  const float* Wo      = (const float*)d_in[9];

  char* ws = (char*)d_ws;
  bf16_t* w1t  = (bf16_t*)(ws + 0);         // 128 KB
  bf16_t* w2t  = (bf16_t*)(ws + 131072);    // 128 KB
  bf16_t* wob  = (bf16_t*)(ws + 262144);    // 2 MB
  bf16_t* qbf  = (bf16_t*)(ws + 2359296);   // 4 MB
  bf16_t* kbf  = (bf16_t*)(ws + 6553600);   // 4 MB
  bf16_t* vtbf = (bf16_t*)(ws + 10747904);  // 4 MB
  float*  attn = (float*)(ws + 14942208);   // 8 MB

  float* out    = (float*)d_out;
  float* out_lq = out + 2097152;
  float* out_lk = out_lq + 524288;
  float* out_lv = out_lk + 524288;
  float* out_rq = out_lv + 524288;
  float* out_rk = out_rq + 524288;
  float* out_rv = out_rk + 524288;

  prep_kernel<<<4096, 256, 0, stream>>>(W1, W2, Wo, w1t, w2t, wob);
  compose_kernel<<<2048, 32, 0, stream>>>(x, proto_q, gate_q, proto_k, gate_k, proto_v,
                                          gate_v, w1t, w2t, out_lq, out_lk, out_lv,
                                          out_rq, out_rk, out_rv, qbf, kbf, vtbf);
  attn_kernel<<<dim3(32, 64), 32, 0, stream>>>(qbf, kbf, vtbf, attn);
  proj_kernel<<<dim3(128, 16), 32, 0, stream>>>(attn, wob, out);
}
